// Attention_15556371546220
// MI455X (gfx1250) — matrix operand order repaired and checked by reading, not compiled, not run on hardware
//
#include <hip/hip_runtime.h>

typedef __attribute__((ext_vector_type(16))) __bf16 v16bf;
typedef __attribute__((ext_vector_type(8)))  float  v8f;
typedef __attribute__((ext_vector_type(4)))  int    v4i;

#define LL   4096
#define CC   768
#define HH   4
#define DKK  64
#define DVV  192
#define M2   8192   // 2L-1 = 8191 pos rows, padded to 8192 (row 8191 = zeros)

#if defined(__gfx1250__) && __has_builtin(__builtin_amdgcn_global_load_async_to_lds_b128)
#define HAVE_ASYNC_LDS 1
#else
#define HAVE_ASYNC_LDS 0
#endif

// ---------------- small helpers ----------------

__device__ __forceinline__ __bf16 f2bf(float f) {
  unsigned u = __builtin_bit_cast(unsigned, f);
  unsigned r = u + 0x7FFFu + ((u >> 16) & 1u);  // round-to-nearest-even
  unsigned short h = (unsigned short)(r >> 16);
  return __builtin_bit_cast(__bf16, h);
}
__device__ __forceinline__ float bf2f(__bf16 b) {
  unsigned short h = __builtin_bit_cast(unsigned short, b);
  unsigned u = ((unsigned)h) << 16;
  return __builtin_bit_cast(float, u);
}
__device__ __forceinline__ v8f v8f_zero() {
  v8f a = {0.f, 0.f, 0.f, 0.f, 0.f, 0.f, 0.f, 0.f};
  return a;
}

// async 16B global->LDS copy (ASYNCcnt path) with synchronous fallback
__device__ __forceinline__ void cp_b128(void* lds, const void* g) {
#if HAVE_ASYNC_LDS
  typedef __attribute__((address_space(1))) v4i* gp_v4i;
  typedef __attribute__((address_space(3))) v4i* lp_v4i;
  __builtin_amdgcn_global_load_async_to_lds_b128((gp_v4i)(void*)g, (lp_v4i)lds, 0, 0);
#else
  *(float4*)lds = *(const float4*)g;
#endif
}
__device__ __forceinline__ void wait_async0() {
#if HAVE_ASYNC_LDS
#if __has_builtin(__builtin_amdgcn_s_wait_asynccnt)
  __builtin_amdgcn_s_wait_asynccnt(0);
#else
  asm volatile("s_wait_asynccnt 0x0" ::: "memory");
#endif
#endif
}

// A-fragment (16x32 bf16, row-major source, leading dim ld):
// lane&15 = row; chunks at K = 8*(lane>>4) and 16+8*(lane>>4)
__device__ __forceinline__ v16bf load_afrag(const __bf16* base, int ld, int lane) {
  const int row = lane & 15, hi = lane >> 4;
  const __bf16* p = base + (size_t)row * ld + 8 * hi;
  v16bf r;
  ((float4*)&r)[0] = *(const float4*)(p);
  ((float4*)&r)[1] = *(const float4*)(p + 16);
  return r;
}
// B-fragment (32x16 bf16) loaded from the TRANSPOSED operand stored row-major
// (rows = N-dim, contiguous K): lane&15 = col(N); chunks at K = 8*(lane>>4) and 16+8*(lane>>4)
__device__ __forceinline__ v16bf load_bfrag(const __bf16* base, int ld, int lane) {
  const int col = lane & 15, hi = lane >> 4;
  const __bf16* p = base + (size_t)col * ld + 8 * hi;
  v16bf r;
  ((float4*)&r)[0] = *(const float4*)(p);
  ((float4*)&r)[1] = *(const float4*)(p + 16);
  return r;
}
__device__ __forceinline__ v8f wmma_bf(v16bf a, v16bf b, v8f c) {
  return __builtin_amdgcn_wmma_f32_16x16x32_bf16(false, a, false, b, (short)0, c, false, false);
}

// ---------------- prep kernels ----------------

__global__ void cvt_bf16_kernel(const float* __restrict__ src, __bf16* __restrict__ dst, int n) {
  for (int i = blockIdx.x * blockDim.x + threadIdx.x; i < n; i += gridDim.x * blockDim.x)
    dst[i] = f2bf(src[i]);
}

// src is (R x N) row-major; dst is (N x R): dst[n*R + r] = src[r*N + n]
__global__ void xpose_bf16_kernel(const float* __restrict__ src, __bf16* __restrict__ dst, int R, int N) {
  int total = R * N;
  for (int i = blockIdx.x * blockDim.x + threadIdx.x; i < total; i += gridDim.x * blockDim.x) {
    int nn = i / R, rr = i - nn * R;
    dst[i] = f2bf(src[rr * N + nn]);
  }
}

// central-mask position features: (8192 x 64) bf16, row 8191 zeroed
__global__ void posmat_kernel(__bf16* __restrict__ dst) {
  const float kk = __logf(4097.0f) * (1.0f / 32.0f);
  const int total = M2 * 64;
  for (int i = blockIdx.x * blockDim.x + threadIdx.x; i < total; i += gridDim.x * blockDim.x) {
    int m = i >> 6, f = i & 63;
    float v = 0.0f;
    if (m < M2 - 1) {
      float p = (float)(m - (LL - 1));
      int ff = f & 31;
      float cw = __expf(kk * (float)(ff + 1)) - 1.0f;
      float emb = (cw > fabsf(p)) ? 1.0f : 0.0f;
      float sg = (p > 0.f) ? 1.f : ((p < 0.f) ? -1.f : 0.f);
      v = (f < 32) ? emb : sg * emb;
    }
    dst[i] = f2bf(v);
  }
}

// ---------------- projection kernels (WMMA) ----------------

// z==0: Qc = x@Wq/8 + cb, Qp = x@Wq/8 + pb (layout [h][n][dk]); z==1: K = x@Wk
__global__ __launch_bounds__(32) void proj_qk_kernel(
    const __bf16* __restrict__ xb, const __bf16* __restrict__ Wqt, const __bf16* __restrict__ Wkt,
    const float* __restrict__ cb, const float* __restrict__ pb,
    __bf16* __restrict__ Qc, __bf16* __restrict__ Qp, __bf16* __restrict__ Kb) {
  const int lane = threadIdx.x & 31, lo = lane & 15, hi = lane >> 4;
  const int n0 = blockIdx.x * 16, j0 = blockIdx.y * 16, z = blockIdx.z;
  const __bf16* A = xb + (size_t)n0 * CC;
  const __bf16* Bt = (z == 0 ? Wqt : Wkt) + (size_t)j0 * CC;
  v8f acc = v8f_zero();
  for (int k = 0; k < CC; k += 32)
    acc = wmma_bf(load_afrag(A + k, CC, lane), load_bfrag(Bt + k, CC, lane), acc);
  const int jj = j0 + lo, h = jj >> 6, d = jj & 63;
  if (z == 0) {
    const float cbv = cb[h * DKK + d], pbv = pb[h * DKK + d];
#pragma unroll
    for (int r = 0; r < 8; ++r) {
      int n = n0 + r + 8 * hi;
      float q = acc[r] * 0.125f;  // 1/sqrt(64)
      Qc[((size_t)h * LL + n) * DKK + d] = f2bf(q + cbv);
      Qp[((size_t)h * LL + n) * DKK + d] = f2bf(q + pbv);
    }
  } else {
#pragma unroll
    for (int r = 0; r < 8; ++r) {
      int n = n0 + r + 8 * hi;
      Kb[((size_t)h * LL + n) * DKK + d] = f2bf(acc[r]);
    }
  }
}

// Vt[h][d][n] = (x@Wv)[n][h*192+d]  (transposed store for contiguous B-frags)
__global__ __launch_bounds__(32) void proj_v_kernel(
    const __bf16* __restrict__ xb, const __bf16* __restrict__ Wvt, __bf16* __restrict__ Vt) {
  const int lane = threadIdx.x & 31, lo = lane & 15, hi = lane >> 4;
  const int n0 = blockIdx.x * 16, j0 = blockIdx.y * 16;
  const __bf16* A = xb + (size_t)n0 * CC;
  const __bf16* Bt = Wvt + (size_t)j0 * CC;
  v8f acc = v8f_zero();
  for (int k = 0; k < CC; k += 32)
    acc = wmma_bf(load_afrag(A + k, CC, lane), load_bfrag(Bt + k, CC, lane), acc);
  const int jj = j0 + lo, h = jj / DVV, d = jj - h * DVV;
#pragma unroll
  for (int r = 0; r < 8; ++r) {
    int n = n0 + r + 8 * hi;
    Vt[((size_t)h * DVV + d) * LL + n] = f2bf(acc[r]);
  }
}

// PK[h][m][dk] = (positions @ Wpos)[m][h*64+d];  m in [0,8191], row 8191 = 0
__global__ __launch_bounds__(32) void posk_kernel(
    const __bf16* __restrict__ posm, const __bf16* __restrict__ Wpost, __bf16* __restrict__ PK) {
  const int lane = threadIdx.x & 31, lo = lane & 15, hi = lane >> 4;
  const int m0 = blockIdx.x * 16, j0 = blockIdx.y * 16;
  const __bf16* A = posm + (size_t)m0 * 64;
  const __bf16* Bt = Wpost + (size_t)j0 * 64;
  v8f acc = v8f_zero();
  acc = wmma_bf(load_afrag(A, 64, lane), load_bfrag(Bt, 64, lane), acc);
  acc = wmma_bf(load_afrag(A + 32, 64, lane), load_bfrag(Bt + 32, 64, lane), acc);
  const int jj = j0 + lo, h = jj >> 6, d = jj & 63;
#pragma unroll
  for (int r = 0; r < 8; ++r) {
    int m = m0 + r + 8 * hi;
    PK[((size_t)h * M2 + m) * DKK + d] = f2bf(acc[r]);
  }
}

// ---------------- fused flash attention with relative shift ----------------
// shifted_pos[i,j] = (Qp_i . PK[(L-1)+j-i]); Toeplitz slice per tile, gathered via LDS.
// K/Vt blocks staged once per workgroup into LDS via async global->LDS copies.
__global__ __launch_bounds__(128) void flash_kernel(
    const __bf16* __restrict__ Qc, const __bf16* __restrict__ Qp,
    const __bf16* __restrict__ Kb, const __bf16* __restrict__ Vt,
    const __bf16* __restrict__ PK, __bf16* __restrict__ AO) {
  __shared__ __align__(16) __bf16 sK[64 * 64];     // K block  [j][dk]
  __shared__ __align__(16) __bf16 sV[192 * 64];    // V block  [d][j]  (transposed)
  __shared__ __align__(16) __bf16 sPre[4][16 * 80];// per-wave Toeplitz slice (col = 15+j-i)
  __shared__ __align__(16) __bf16 sP[4][16 * 64];  // per-wave probabilities (D->A relayout)
  const int tid = threadIdx.x;
  const int lane = tid & 31, w = tid >> 5;
  const int lo = lane & 15, hi = lane >> 4;
  const int h = blockIdx.y;
  const int iw = (blockIdx.x * 4 + w) * 16;  // this wave's query-row base
  __bf16* myPre = sPre[w];
  __bf16* myP = sP[w];

  const __bf16* qcb = Qc + ((size_t)h * LL + iw) * DKK;
  const __bf16* qpb = Qp + ((size_t)h * LL + iw) * DKK;
  const v16bf aQc0 = load_afrag(qcb, DKK, lane);
  const v16bf aQc1 = load_afrag(qcb + 32, DKK, lane);
  const v16bf aQp0 = load_afrag(qpb, DKK, lane);
  const v16bf aQp1 = load_afrag(qpb + 32, DKK, lane);

  v8f accO[12];
#pragma unroll
  for (int t = 0; t < 12; ++t) accO[t] = v8f_zero();
  float rmax[8], rsum[8];
#pragma unroll
  for (int r = 0; r < 8; ++r) { rmax[r] = -3.0e38f; rsum[r] = 0.0f; }

  const __bf16* kh = Kb + (size_t)h * LL * DKK;
  const __bf16* vh = Vt + (size_t)h * DVV * LL;

  for (int j0 = 0; j0 < LL; j0 += 64) {
    // ---- stage K (64x64) and Vt (192x64) blocks into LDS (async path) ----
    __syncthreads();  // prior iteration done reading sK/sV
    {
      const __bf16* kb = kh + (size_t)j0 * DKK;  // contiguous 64*64 block
#pragma unroll
      for (int t = 0; t < 4; ++t) {
        int idx = tid + t * 128;                 // 512 16B-chunks
        cp_b128(sK + idx * 8, kb + (size_t)idx * 8);
      }
#pragma unroll
      for (int t = 0; t < 12; ++t) {
        int idx = tid + t * 128;                 // 1536 16B-chunks (192 rows x 8)
        int d = idx >> 3, c = idx & 7;
        cp_b128(sV + d * 64 + c * 8, vh + (size_t)d * LL + j0 + c * 8);
      }
    }
    // prefetch next iteration's per-wave PK slice while async copies fly
    const int mb = (LL - 1) + j0 - iw - 15;
    if (j0 + 64 < LL) {
      const __bf16* pkn = PK + ((size_t)h * M2 + (mb + 64)) * DKK;
      __builtin_prefetch(pkn + (size_t)lane * 2 * DKK, 0, 2);
      __builtin_prefetch(pkn + (size_t)(lane * 2 + 1) * DKK, 0, 2);
      __builtin_prefetch(pkn + (size_t)(64 + (lane & 15)) * DKK, 0, 2);
    }
    wait_async0();
    __syncthreads();  // sK/sV filled

    // ---- content logits: Qc(16x64) @ K_block^T (64x64), B-frags from LDS ----
    v8f S[4];
#pragma unroll
    for (int ct = 0; ct < 4; ++ct) {
      v8f a = v8f_zero();
      a = wmma_bf(aQc0, load_bfrag(sK + ct * 16 * DKK, DKK, lane), a);
      a = wmma_bf(aQc1, load_bfrag(sK + ct * 16 * DKK + 32, DKK, lane), a);
      S[ct] = a;
    }
    // ---- pos-logit slice: Qp(16x64) @ PK[mb .. mb+79]^T (global, per-wave) ----
    const __bf16* pkb = PK + ((size_t)h * M2 + mb) * DKK;
#pragma unroll
    for (int pt = 0; pt < 5; ++pt) {
      v8f a = v8f_zero();
      a = wmma_bf(aQp0, load_bfrag(pkb + (size_t)pt * 16 * DKK, DKK, lane), a);
      a = wmma_bf(aQp1, load_bfrag(pkb + (size_t)pt * 16 * DKK + 32, DKK, lane), a);
#pragma unroll
      for (int r = 0; r < 8; ++r)
        myPre[(r + 8 * hi) * 80 + pt * 16 + lo] = f2bf(a[r]);
    }
    __builtin_amdgcn_wave_barrier();
    // ---- relative-shift gather: S[i][j] += Pre[i][15 + j - i] ----
#pragma unroll
    for (int ct = 0; ct < 4; ++ct)
#pragma unroll
      for (int r = 0; r < 8; ++r) {
        int i = r + 8 * hi, j = ct * 16 + lo;
        S[ct][r] += bf2f(myPre[i * 80 + 15 + j - i]);
      }
    // ---- online softmax ----
    float sc[8];
#pragma unroll
    for (int r = 0; r < 8; ++r) {
      float m = fmaxf(fmaxf(S[0][r], S[1][r]), fmaxf(S[2][r], S[3][r]));
      m = fmaxf(m, __shfl_xor(m, 1));
      m = fmaxf(m, __shfl_xor(m, 2));
      m = fmaxf(m, __shfl_xor(m, 4));
      m = fmaxf(m, __shfl_xor(m, 8));
      float nm = fmaxf(rmax[r], m);
      sc[r] = __expf(rmax[r] - nm);
      rmax[r] = nm;
      rsum[r] *= sc[r];
    }
#pragma unroll
    for (int t = 0; t < 12; ++t)
#pragma unroll
      for (int r = 0; r < 8; ++r) accO[t][r] *= sc[r];
    float ps[8];
#pragma unroll
    for (int r = 0; r < 8; ++r) ps[r] = 0.f;
#pragma unroll
    for (int ct = 0; ct < 4; ++ct)
#pragma unroll
      for (int r = 0; r < 8; ++r) {
        float p = __expf(S[ct][r] - rmax[r]);
        ps[r] += p;
        myP[(r + 8 * hi) * 64 + ct * 16 + lo] = f2bf(p);
      }
#pragma unroll
    for (int r = 0; r < 8; ++r) {
      float s = ps[r];
      s += __shfl_xor(s, 1);
      s += __shfl_xor(s, 2);
      s += __shfl_xor(s, 4);
      s += __shfl_xor(s, 8);
      rsum[r] += s;
    }
    __builtin_amdgcn_wave_barrier();
    // ---- O += P(16x64) @ V_block(64x192), B-frags from LDS sV ----
    const v16bf aP0 = load_afrag(myP, 64, lane);
    const v16bf aP1 = load_afrag(myP + 32, 64, lane);
#pragma unroll
    for (int t = 0; t < 12; ++t) {
      accO[t] = wmma_bf(aP0, load_bfrag(sV + t * 16 * 64, 64, lane), accO[t]);
      accO[t] = wmma_bf(aP1, load_bfrag(sV + t * 16 * 64 + 32, 64, lane), accO[t]);
    }
  }
  // ---- epilogue: normalize, store AO[n][h*192+d] as bf16 ----
#pragma unroll
  for (int r = 0; r < 8; ++r) rsum[r] = 1.0f / rsum[r];
#pragma unroll
  for (int t = 0; t < 12; ++t)
#pragma unroll
    for (int r = 0; r < 8; ++r) {
      int n = iw + r + 8 * hi, d = t * 16 + lo;
      AO[(size_t)n * (HH * DVV) + h * DVV + d] = f2bf(accO[t][r] * rsum[r]);
    }
}

// ---------------- output projection: out = AO @ Wout + bout (f32) ----------------
__global__ __launch_bounds__(32) void outproj_kernel(
    const __bf16* __restrict__ AO, const __bf16* __restrict__ Woutt,
    const float* __restrict__ bout, float* __restrict__ out) {
  const int lane = threadIdx.x & 31, lo = lane & 15, hi = lane >> 4;
  const int n0 = blockIdx.x * 16, c0 = blockIdx.y * 16;
  const __bf16* A = AO + (size_t)n0 * CC;
  const __bf16* Bt = Woutt + (size_t)c0 * CC;
  v8f acc = v8f_zero();
  for (int k = 0; k < CC; k += 32)
    acc = wmma_bf(load_afrag(A + k, CC, lane), load_bfrag(Bt + k, CC, lane), acc);
  const int c = c0 + lo;
  const float b = bout[c];
#pragma unroll
  for (int r = 0; r < 8; ++r) {
    int n = n0 + r + 8 * hi;
    out[(size_t)n * CC + c] = acc[r] + b;
  }
}

// ---------------- host launcher ----------------

extern "C" void kernel_launch(void* const* d_in, const int* in_sizes, int n_in,
                              void* d_out, int out_size, void* d_ws, size_t ws_size,
                              hipStream_t stream) {
  const float* x    = (const float*)d_in[0];
  const float* Wq   = (const float*)d_in[1];
  const float* Wk   = (const float*)d_in[2];
  const float* Wv   = (const float*)d_in[3];
  const float* Wpos = (const float*)d_in[4];
  const float* Wout = (const float*)d_in[5];
  const float* bout = (const float*)d_in[6];
  const float* cb   = (const float*)d_in[7];
  const float* pb   = (const float*)d_in[8];

  char* ws = (char*)d_ws;
  size_t off = 0;
  auto alloc = [&](size_t elems) {
    void* p = ws + off;
    off += (elems * sizeof(__bf16) + 255) & ~(size_t)255;
    return (__bf16*)p;
  };
  __bf16* xb    = alloc((size_t)LL * CC);
  __bf16* Wqt   = alloc((size_t)(HH * DKK) * CC);
  __bf16* Wkt   = alloc((size_t)(HH * DKK) * CC);
  __bf16* Wvt   = alloc((size_t)(HH * DVV) * CC);
  __bf16* Wpost = alloc((size_t)(HH * DKK) * 64);
  __bf16* Woutt = alloc((size_t)CC * CC);
  __bf16* posm  = alloc((size_t)M2 * 64);
  __bf16* Qc    = alloc((size_t)HH * LL * DKK);
  __bf16* Qp    = alloc((size_t)HH * LL * DKK);
  __bf16* Kb    = alloc((size_t)HH * LL * DKK);
  __bf16* Vt    = alloc((size_t)HH * DVV * LL);
  __bf16* PK    = alloc((size_t)HH * M2 * DKK);
  __bf16* AO    = alloc((size_t)LL * (HH * DVV));

  // prep: bf16 conversions / transposes / position features
  cvt_bf16_kernel<<<2048, 256, 0, stream>>>(x, xb, LL * CC);
  xpose_bf16_kernel<<<512, 256, 0, stream>>>(Wq, Wqt, CC, HH * DKK);
  xpose_bf16_kernel<<<512, 256, 0, stream>>>(Wk, Wkt, CC, HH * DKK);
  xpose_bf16_kernel<<<1024, 256, 0, stream>>>(Wv, Wvt, CC, HH * DVV);
  xpose_bf16_kernel<<<64, 256, 0, stream>>>(Wpos, Wpost, 64, HH * DKK);
  xpose_bf16_kernel<<<1024, 256, 0, stream>>>(Wout, Woutt, CC, CC);
  posmat_kernel<<<1024, 256, 0, stream>>>(posm);

  // projections (WMMA)
  proj_qk_kernel<<<dim3(LL / 16, (HH * DKK) / 16, 2), 32, 0, stream>>>(
      xb, Wqt, Wkt, cb, pb, Qc, Qp, Kb);
  proj_v_kernel<<<dim3(LL / 16, (HH * DVV) / 16), 32, 0, stream>>>(xb, Wvt, Vt);
  posk_kernel<<<dim3(M2 / 16, (HH * DKK) / 16), 32, 0, stream>>>(posm, Wpost, PK);

  // fused attention (WMMA flash + relative shift, async LDS staging)
  flash_kernel<<<dim3(LL / 64, HH), 128, 0, stream>>>(Qc, Qp, Kb, Vt, PK, AO);

  // output projection (WMMA)
  outproj_kernel<<<dim3(LL / 16, CC / 16), 32, 0, stream>>>(AO, Woutt, bout, (float*)d_out);
}
